// SoftDTW_77635828843187
// MI455X (gfx1250) — compile-verified, matrix operand order checked by reading, not run on hardware
//
#include <hip/hip_runtime.h>
#include <hip/hip_bf16.h>

typedef __attribute__((ext_vector_type(16))) _Float16 v16h;
typedef __attribute__((ext_vector_type(8)))  float    v8f;

#define SDTW_LARGE 1e30f
#define NSEQ   512
#define DFEAT  64
#define NBATCH 128
#define NDIAG  (2 * NSEQ - 1)   // 1023

// Load 16 f32 (two 32B segments) from one row, split each into f16 hi/lo,
// and accumulate the squared-norm partial for this lane's row/col.
__device__ __forceinline__ void load_frag_split(const float* __restrict__ row,
                                                int kbase, v16h& hi, v16h& lo,
                                                float& sq) {
  const float4* p0 = reinterpret_cast<const float4*>(row + kbase);
  const float4* p1 = reinterpret_cast<const float4*>(row + kbase + 16);
  float4 q0 = p0[0], q1 = p0[1], q2 = p1[0], q3 = p1[1];
  float v[16] = {q0.x, q0.y, q0.z, q0.w, q1.x, q1.y, q1.z, q1.w,
                 q2.x, q2.y, q2.z, q2.w, q3.x, q3.y, q3.z, q3.w};
#pragma unroll
  for (int e = 0; e < 16; ++e) {
    float f = v[e];
    sq += f * f;
    _Float16 h = (_Float16)f;
    hi[e] = h;
    lo[e] = (_Float16)(f - (float)h);
  }
}

// Kernel 1: per-wave 16x16 tile of D[b] = x2 + y2 - 2*X.Y^T via split-f16 WMMA.
// Grid: (128 tiles-of-8, nb batches), block 256 (8 waves).
__global__ void __launch_bounds__(256)
pairdist_wmma_kernel(const float* __restrict__ X, const float* __restrict__ Y,
                     float* __restrict__ D, int b0) {
  const int lane = threadIdx.x & 31;
  const int wave = threadIdx.x >> 5;
  const int tile = blockIdx.x * 8 + wave;       // 0..1023
  const int ti = tile >> 5, tj = tile & 31;
  const int b  = b0 + blockIdx.y;
  const int I0 = ti * 16, J0 = tj * 16;
  const int r    = lane & 15;
  const int koff = (lane >> 4) << 3;            // 0 (lanes 0-15) or 8 (16-31)

  const float* xrow = X + ((size_t)b * NSEQ + (I0 + r)) * DFEAT;
  const float* yrow = Y + ((size_t)b * NSEQ + (J0 + r)) * DFEAT;

  v8f acc = {};
  float pA = 0.f, pB = 0.f;
#pragma unroll
  for (int kc = 0; kc < DFEAT; kc += 32) {
    v16h ahi, alo, bhi, blo;
    load_frag_split(xrow, kc + koff, ahi, alo, pA);
    load_frag_split(yrow, kc + koff, bhi, blo, pB);
    // dot ~= hi.hi + hi.lo + lo.hi  (fp32-accurate split product)
    acc = __builtin_amdgcn_wmma_f32_16x16x32_f16(false, ahi, false, bhi,
                                                 (short)0, acc, false, false);
    acc = __builtin_amdgcn_wmma_f32_16x16x32_f16(false, ahi, false, blo,
                                                 (short)0, acc, false, false);
    acc = __builtin_amdgcn_wmma_f32_16x16x32_f16(false, alo, false, bhi,
                                                 (short)0, acc, false, false);
  }

  // lanes l and l^16 jointly cover all 64 K values of row/col (l&15)
  float x2 = pA + __shfl_xor(pA, 16, 32);   // ||x[I0+r]||^2
  float y2 = pB + __shfl_xor(pB, 16, 32);   // ||y[J0+r]||^2

  float* dbase = D + (size_t)blockIdx.y * NSEQ * NSEQ;
#pragma unroll
  for (int v = 0; v < 8; ++v) {
    int M = v + koff;                        // C/D layout: M = v (+8 for hi lanes)
    float x2v = __shfl(x2, M, 32);           // lane M holds row I0+M norm
    float dv  = x2v + y2 - 2.0f * acc[v];
    dbase[(size_t)(I0 + M) * NSEQ + (J0 + r)] = dv;
  }
}

// Kernel 2: soft-DTW anti-diagonal scan. One block per batch, one thread per
// row. D row segments staged into registers 64 columns at a time; three LDS
// diagonal buffers; one workgroup barrier per step.
__global__ void __launch_bounds__(512)
softdtw_scan_kernel(const float* __restrict__ D, float* __restrict__ out, int b0) {
  __shared__ float rbuf[3][NSEQ + 1];        // slot 0 = LARGE sentinel (i-1 edge)
  const int i   = threadIdx.x;               // row index 0..511
  const int blk = blockIdx.x;
  const float* drow = D + ((size_t)blk * NSEQ + i) * NSEQ;

  rbuf[0][i + 1] = SDTW_LARGE;
  rbuf[1][i + 1] = SDTW_LARGE;
  rbuf[2][i + 1] = SDTW_LARGE;
  if (i == 0) { rbuf[0][0] = SDTW_LARGE; rbuf[1][0] = SDTW_LARGE; rbuf[2][0] = SDTW_LARGE; }
  __syncthreads();

  float* cu  = rbuf[0];   // diag p
  float* pv  = rbuf[1];   // diag p-1
  float* pv2 = rbuf[2];   // diag p-2

  for (int p0 = 0; p0 < NDIAG; p0 += 64) {
    // Stage this thread's D[i, p0-i .. p0-i+63] (clamped) into registers:
    // contiguous per-thread addresses, consumed one per step.
    float dreg[64];
    const int jb = p0 - i;
#pragma unroll
    for (int c = 0; c < 64; ++c) {
      int j = jb + c;
      j = (j < 0) ? 0 : (j > NSEQ - 1 ? NSEQ - 1 : j);
      dreg[c] = drow[j];
    }
#pragma unroll
    for (int c = 0; c < 64; ++c) {
      const int p = p0 + c;                  // uniform across block
      if (p < NDIAG) {
        const int j = p - i;
        const bool valid = (j >= 0) && (j < NSEQ);
        float r_nw = pv2[i];                 // R[i-1][j-1]
        float r_n  = pv[i];                  // R[i-1][j]
        float r_w  = pv[i + 1];              // R[i][j-1]
        if (p == 0 && i == 0) r_nw = 0.0f;   // DP origin
        // softmin with gamma = 1
        float r0 = -r_nw, r1 = -r_n, r2 = -r_w;
        float rmax = fmaxf(fmaxf(r0, r1), r2);
        float rsum = __expf(r0 - rmax) + __expf(r1 - rmax) + __expf(r2 - rmax);
        float softmin = -(__logf(rsum) + rmax);
        float cur = valid ? (dreg[c] + softmin) : SDTW_LARGE;
        cu[i + 1] = cur;
        if (p == NDIAG - 1 && i == NSEQ - 1) out[b0 + blk] = cur;
        __syncthreads();
        float* t = pv2; pv2 = pv; pv = cu; cu = t;  // rotate diagonals
      }
    }
  }
}

extern "C" void kernel_launch(void* const* d_in, const int* in_sizes, int n_in,
                              void* d_out, int out_size, void* d_ws, size_t ws_size,
                              hipStream_t stream) {
  const float* x = (const float*)d_in[0];
  const float* y = (const float*)d_in[1];
  float* out = (float*)d_out;
  float* ws  = (float*)d_ws;

  // D workspace: 512*512*4 B per batch (128 MiB total). If ws is smaller,
  // process batches in chunks (deterministic, capture-safe).
  const size_t perBatchBytes = (size_t)NSEQ * NSEQ * sizeof(float);
  size_t maxb = ws_size / perBatchBytes;
  int bc = (int)(maxb < 1 ? 1 : (maxb > (size_t)NBATCH ? (size_t)NBATCH : maxb));

  for (int bStart = 0; bStart < NBATCH; bStart += bc) {
    int nb = (NBATCH - bStart < bc) ? (NBATCH - bStart) : bc;
    dim3 g1(128, nb);                         // 1024 16x16 tiles per batch / 8 waves
    pairdist_wmma_kernel<<<g1, 256, 0, stream>>>(x, y, ws, bStart);
    softdtw_scan_kernel<<<nb, 512, 0, stream>>>(ws, out, bStart);
  }
}
